// RNN_13597866459300
// MI455X (gfx1250) — matrix operand order repaired and checked by reading, not compiled, not run on hardware
//
#include <hip/hip_runtime.h>
#include <stdint.h>

// Problem constants
#define B_   4096
#define T_   5
#define E_   4800
#define H_   1000
#define HP_  1024                 // H padded to multiple of 32 (K padding)
#define KG_  (E_ + HP_)           // 5824 : padded K for both GEMMs
#define NG_  2048                 // padded N for gates (2*H -> 2048)
#define NC_  1024                 // padded N for cand  (H   -> 1024)
#define KT_  (KG_ / 32)           // 182 K-steps
#define KTX_ (E_ / 32)            // 150 K-steps over the x segment

typedef __attribute__((ext_vector_type(16))) __bf16        v16bf;
typedef __attribute__((ext_vector_type(8)))  float         v8f;
typedef __attribute__((ext_vector_type(4)))  unsigned int  u32x4;

union FragU { v16bf bf; u32x4 u[2]; };

// ---------------------------------------------------------------------------
// A-fragment (16x32 bf16, MxK) via a precomputed per-lane pointer.
// Per ISA 7.12.2: lane 0-15: M=lane, K 0..7 / 16..23; lane 16-31: K 8..15 / 24..31.
// Pointer already includes row*stride + laneHi*8 elements.
__device__ __forceinline__ v16bf load_fragA_p(const __bf16* p) {
  FragU f;
  f.u[0] = *(const u32x4*)p;
  f.u[1] = *(const u32x4*)(p + 16);
  return f.bf;
}

// B-fragment (32x16 bf16, KxN) from the LDS-staged tile (128 rows x 64B).
// Layout: lanes 0-15 hold K=0..7 / 16..23 (N=lane), lanes 16-31 hold K=8..15 / 24..31.
__device__ __forceinline__ v16bf load_fragB_lds(const __bf16* tile,
                                                int row, int lane) {
  const __bf16* q = tile + (row + (lane & 15)) * 32 + ((lane >> 4) << 3);
  FragU f;
  f.u[0] = *(const u32x4*)q;
  f.u[1] = *(const u32x4*)(q + 16);
  return f.bf;
}

// Async-stage one 128x32 bf16 weight tile (rows nblock.., cols k0..k0+31 of Wt)
// into LDS at lds_base. 8 waves x 2 issues, each lane 16B. ASYNCcnt-tracked.
__device__ __forceinline__ void async_stage_B(const __bf16* __restrict__ Wt,
                                              int nblock, int k0,
                                              uint32_t lds_base,
                                              int wave, int lane) {
  const int r0    = wave * 16 + (lane >> 2);   // 8 rows per issue
  const int chunk = lane & 3;                  // 4 x 16B per 64B row
  #pragma unroll
  for (int s = 0; s < 2; ++s) {
    const int row = r0 + s * 8;
    const unsigned long long ga =
        (unsigned long long)(uintptr_t)(Wt + (size_t)(nblock + row) * KG_ + k0)
        + (unsigned long long)(chunk * 16);
    const uint32_t la = lds_base + (uint32_t)(row * 64 + chunk * 16);
    asm volatile("global_load_async_to_lds_b128 %0, %1, off"
                 :: "v"(la), "v"(ga) : "memory");
  }
}

__device__ __forceinline__ void wait_async0() {
  asm volatile("s_wait_asynccnt 0" ::: "memory");
}

__device__ __forceinline__ float sigmoidf_(float x) {
  return 1.0f / (1.0f + __expf(-x));
}

#define WMMA16(A0, A1, A2, A3, B0, B1, B2, B3)                                              \
  do {                                                                                       \
    acc[0][0] = __builtin_amdgcn_wmma_f32_16x16x32_bf16(false, A0, false, B0, (short)0, acc[0][0], false, false); \
    acc[0][1] = __builtin_amdgcn_wmma_f32_16x16x32_bf16(false, A0, false, B1, (short)0, acc[0][1], false, false); \
    acc[0][2] = __builtin_amdgcn_wmma_f32_16x16x32_bf16(false, A0, false, B2, (short)0, acc[0][2], false, false); \
    acc[0][3] = __builtin_amdgcn_wmma_f32_16x16x32_bf16(false, A0, false, B3, (short)0, acc[0][3], false, false); \
    acc[1][0] = __builtin_amdgcn_wmma_f32_16x16x32_bf16(false, A1, false, B0, (short)0, acc[1][0], false, false); \
    acc[1][1] = __builtin_amdgcn_wmma_f32_16x16x32_bf16(false, A1, false, B1, (short)0, acc[1][1], false, false); \
    acc[1][2] = __builtin_amdgcn_wmma_f32_16x16x32_bf16(false, A1, false, B2, (short)0, acc[1][2], false, false); \
    acc[1][3] = __builtin_amdgcn_wmma_f32_16x16x32_bf16(false, A1, false, B3, (short)0, acc[1][3], false, false); \
    acc[2][0] = __builtin_amdgcn_wmma_f32_16x16x32_bf16(false, A2, false, B0, (short)0, acc[2][0], false, false); \
    acc[2][1] = __builtin_amdgcn_wmma_f32_16x16x32_bf16(false, A2, false, B1, (short)0, acc[2][1], false, false); \
    acc[2][2] = __builtin_amdgcn_wmma_f32_16x16x32_bf16(false, A2, false, B2, (short)0, acc[2][2], false, false); \
    acc[2][3] = __builtin_amdgcn_wmma_f32_16x16x32_bf16(false, A2, false, B3, (short)0, acc[2][3], false, false); \
    acc[3][0] = __builtin_amdgcn_wmma_f32_16x16x32_bf16(false, A3, false, B0, (short)0, acc[3][0], false, false); \
    acc[3][1] = __builtin_amdgcn_wmma_f32_16x16x32_bf16(false, A3, false, B1, (short)0, acc[3][1], false, false); \
    acc[3][2] = __builtin_amdgcn_wmma_f32_16x16x32_bf16(false, A3, false, B2, (short)0, acc[3][2], false, false); \
    acc[3][3] = __builtin_amdgcn_wmma_f32_16x16x32_bf16(false, A3, false, B3, (short)0, acc[3][3], false, false); \
  } while (0)

// ---------------------------------------------------------------------------
// Fused GEMM: C = [Xb | Hpart] @ Wt^T  (+bias, + GRU epilogue)
// MODE 0: gates — ru = sigmoid(.); writes U (update gate) and RHb = bf16(r*h)
// MODE 1: cand  — c = tanh(.); h' = u*h + (1-u)*c; writes Hf (f32) and Hb (bf16)
// Block tile 256(M) x 128(N), 8 waves (4x2), wave tile 64x64 (4x4 WMMA frags).
// B (weights) double-buffered through LDS via async-to-LDS DMA; A fragments
// loaded direct from global, issued BEFORE the barrier so their latency hides
// behind the async wait.
template <int MODE>
__global__ __launch_bounds__(256)
void gru_gemm(const __bf16* __restrict__ Xb,     // B x E   (bf16)
              const __bf16* __restrict__ Hpart,  // B x HP  (Hb gates / RHb cand)
              const __bf16* __restrict__ Wt,     // Npad x KG (bf16, transposed)
              const float*  __restrict__ bias,
              float*  __restrict__ Hf,           // B x HP f32 hidden state
              float*  __restrict__ U,            // B x HP f32 update gate
              __bf16* __restrict__ RHb,          // B x HP bf16 (r*h)
              __bf16* __restrict__ Hb)           // B x HP bf16 (h)
{
  __shared__ __bf16 Bt[2][128][32];              // 2 x 8KB double buffer

  const int lane = threadIdx.x & 31;
  const int wave = threadIdx.x >> 5;
  const int wn   = wave & 1;                     // 2 waves along N
  const int wm   = wave >> 1;                    // 4 waves along M
  const int mbase  = blockIdx.y * 256 + wm * 64;
  const int nblock = blockIdx.x * 128;

  const uint32_t lb0 = (uint32_t)(uintptr_t)&Bt[0][0][0];
  const uint32_t lb1 = (uint32_t)(uintptr_t)&Bt[1][0][0];

  const int laneRow = lane & 15;
  const int laneK8  = (lane >> 4) << 3;          // element offset (K 0 vs 8)

  v8f acc[4][4] = {};

  // Prologue: stage K-step 0 into buffer 0.
  async_stage_B(Wt, nblock, 0, lb0, wave, lane);

  // Persistent per-lane A pointers: x segment (stride E), bump 32 elems/iter.
  const __bf16* a0p = Xb + (size_t)(mbase +  0 + laneRow) * E_ + laneK8;
  const __bf16* a1p = Xb + (size_t)(mbase + 16 + laneRow) * E_ + laneK8;
  const __bf16* a2p = Xb + (size_t)(mbase + 32 + laneRow) * E_ + laneK8;
  const __bf16* a3p = Xb + (size_t)(mbase + 48 + laneRow) * E_ + laneK8;

  for (int kt = 0; kt < KTX_; ++kt) {
    // A loads first: independent of the LDS staging, overlap barrier latency.
    v16bf a0 = load_fragA_p(a0p);
    v16bf a1 = load_fragA_p(a1p);
    v16bf a2 = load_fragA_p(a2p);
    v16bf a3 = load_fragA_p(a3p);
    a0p += 32; a1p += 32; a2p += 32; a3p += 32;

    wait_async0();
    __syncthreads();
    async_stage_B(Wt, nblock, (kt + 1) * 32, (kt & 1) ? lb0 : lb1, wave, lane);

    const __bf16* tile = (kt & 1) ? &Bt[1][0][0] : &Bt[0][0][0];
    v16bf b0 = load_fragB_lds(tile, wn * 64 +  0, lane);
    v16bf b1 = load_fragB_lds(tile, wn * 64 + 16, lane);
    v16bf b2 = load_fragB_lds(tile, wn * 64 + 32, lane);
    v16bf b3 = load_fragB_lds(tile, wn * 64 + 48, lane);

    WMMA16(a0, a1, a2, a3, b0, b1, b2, b3);
  }

  // Recurrent segment (stride HP).
  a0p = Hpart + (size_t)(mbase +  0 + laneRow) * HP_ + laneK8;
  a1p = Hpart + (size_t)(mbase + 16 + laneRow) * HP_ + laneK8;
  a2p = Hpart + (size_t)(mbase + 32 + laneRow) * HP_ + laneK8;
  a3p = Hpart + (size_t)(mbase + 48 + laneRow) * HP_ + laneK8;

  for (int kt = KTX_; kt < KT_; ++kt) {
    v16bf a0 = load_fragA_p(a0p);
    v16bf a1 = load_fragA_p(a1p);
    v16bf a2 = load_fragA_p(a2p);
    v16bf a3 = load_fragA_p(a3p);
    a0p += 32; a1p += 32; a2p += 32; a3p += 32;

    wait_async0();
    __syncthreads();
    if (kt + 1 < KT_)
      async_stage_B(Wt, nblock, (kt + 1) * 32, (kt & 1) ? lb0 : lb1, wave, lane);

    const __bf16* tile = (kt & 1) ? &Bt[1][0][0] : &Bt[0][0][0];
    v16bf b0 = load_fragB_lds(tile, wn * 64 +  0, lane);
    v16bf b1 = load_fragB_lds(tile, wn * 64 + 16, lane);
    v16bf b2 = load_fragB_lds(tile, wn * 64 + 32, lane);
    v16bf b3 = load_fragB_lds(tile, wn * 64 + 48, lane);

    WMMA16(a0, a1, a2, a3, b0, b1, b2, b3);
  }

  // ---- epilogue ----
  // C/D layout: lane&15 = N column, (lane>>4)*8 + vgpr = M row per 16x16 tile.
  #pragma unroll
  for (int i = 0; i < 4; ++i) {
    #pragma unroll
    for (int j = 0; j < 4; ++j) {
      const int ncol = nblock + wn * 64 + j * 16 + (lane & 15);
      const int mrow = mbase + i * 16 + ((lane >> 4) << 3);
      if (MODE == 0) {
        if (ncol < 2 * H_) {
          const float bn  = bias[ncol];
          const bool  isR = (ncol < H_);
          const int   nh  = isR ? ncol : (ncol - H_);
          #pragma unroll
          for (int e = 0; e < 8; ++e) {
            const int m = mrow + e;
            const size_t idx = (size_t)m * HP_ + nh;
            const float s = sigmoidf_(acc[i][j][e] + bn);
            if (isR) RHb[idx] = (__bf16)(s * Hf[idx]);   // r * h_{t-1}
            else     U[idx]   = s;                        // update gate u
          }
        }
      } else {
        if (ncol < H_) {
          const float bn = bias[ncol];
          #pragma unroll
          for (int e = 0; e < 8; ++e) {
            const int m = mrow + e;
            const size_t idx = (size_t)m * HP_ + ncol;
            const float c  = tanhf(acc[i][j][e] + bn);
            const float u  = U[idx];
            const float h  = Hf[idx];
            const float hn = u * h + (1.0f - u) * c;
            Hf[idx] = hn;
            Hb[idx] = (__bf16)hn;
          }
        }
      }
    }
  }
}

// ---------------------------------------------------------------------------
// Transpose + f32->bf16 convert: W (Kin x Nin, f32) -> Wt (Npad x KG, bf16),
// zero-padded in both N (rows of Wt) and K (cols of Wt). 32x32 LDS tiles.
__global__ __launch_bounds__(256)
void conv_wt(const float* __restrict__ W, __bf16* __restrict__ Wt,
             int Kin, int Nin)
{
  __shared__ float tile[32][33];
  const int tx = threadIdx.x;   // 0..31
  const int ty = threadIdx.y;   // 0..7
  const int ntile = blockIdx.x * 32;
  const int ktile = blockIdx.y * 32;
  #pragma unroll
  for (int r = 0; r < 4; ++r) {
    const int k = ktile + ty + r * 8;
    const int n = ntile + tx;
    tile[ty + r * 8][tx] = (k < Kin && n < Nin) ? W[(size_t)k * Nin + n] : 0.0f;
  }
  __syncthreads();
  #pragma unroll
  for (int r = 0; r < 4; ++r) {
    const int n = ntile + ty + r * 8;
    const int k = ktile + tx;
    Wt[(size_t)n * KG_ + k] = (__bf16)tile[tx][ty + r * 8];
  }
}

// Convert x_t slice (B x E from inputs[:, t, :]) to bf16.
__global__ __launch_bounds__(256)
void conv_x(const float* __restrict__ inputs, __bf16* __restrict__ Xb, int t)
{
  const int e = blockIdx.x * 256 + threadIdx.x;
  const int b = blockIdx.y;
  if (e < E_)
    Xb[(size_t)b * E_ + e] =
        (__bf16)inputs[(size_t)b * (T_ * E_) + (size_t)t * E_ + e];
}

// Zero a 16B-aligned region (grid-stride).
__global__ __launch_bounds__(256)
void zero_ws(u32x4* __restrict__ p, size_t n16)
{
  const u32x4 z = {0u, 0u, 0u, 0u};
  for (size_t i = blockIdx.x * 256ull + threadIdx.x; i < n16;
       i += (size_t)gridDim.x * 256ull)
    p[i] = z;
}

// Final FC: out[m] = dot(Hf[m, :H], Wfc) + b   (HP pad of Hf is zero).
__global__ __launch_bounds__(256)
void final_fc(const float* __restrict__ Hf, const float* __restrict__ Wfc,
              const float* __restrict__ bfc, float* __restrict__ out)
{
  const int lane = threadIdx.x & 31;
  const int wave = threadIdx.x >> 5;
  const int m = blockIdx.x * 8 + wave;
  float s = 0.0f;
  for (int k = lane; k < H_; k += 32)
    s += Hf[(size_t)m * HP_ + k] * Wfc[k];
  for (int off = 16; off > 0; off >>= 1)
    s += __shfl_down(s, off, 32);
  if (lane == 0) out[m] = s + bfc[0];
}

// ---------------------------------------------------------------------------
extern "C" void kernel_launch(void* const* d_in, const int* in_sizes, int n_in,
                              void* d_out, int out_size, void* d_ws, size_t ws_size,
                              hipStream_t stream)
{
  (void)in_sizes; (void)n_in; (void)out_size; (void)ws_size;
  const float* inputs = (const float*)d_in[0];
  const float* Wg     = (const float*)d_in[1];
  const float* bg     = (const float*)d_in[2];
  const float* Wc     = (const float*)d_in[3];
  const float* bc     = (const float*)d_in[4];
  const float* Wfc    = (const float*)d_in[5];
  const float* bfc    = (const float*)d_in[6];
  float* out = (float*)d_out;

  // Workspace carve-up (256B aligned slices).
  char* ws = (char*)d_ws;
  size_t off = 0;
  auto carve = [&](size_t bytes) -> char* {
    char* p = ws + off;
    off = (off + bytes + 255) & ~(size_t)255;
    return p;
  };
  __bf16* Wgt = (__bf16*)carve((size_t)NG_ * KG_ * 2);   // 23.9 MB
  __bf16* Wct = (__bf16*)carve((size_t)NC_ * KG_ * 2);   // 11.9 MB
  __bf16* Xb  = (__bf16*)carve((size_t)B_ * E_ * 2);     // 39.3 MB
  char* zbase = ws + off;                                 // state region (zeroed)
  __bf16* Hb  = (__bf16*)carve((size_t)B_ * HP_ * 2);    //  8.4 MB
  __bf16* RHb = (__bf16*)carve((size_t)B_ * HP_ * 2);    //  8.4 MB
  float*  Hf  = (float*) carve((size_t)B_ * HP_ * 4);    // 16.8 MB
  float*  U   = (float*) carve((size_t)B_ * HP_ * 4);    // 16.8 MB
  const size_t zbytes = (size_t)((ws + off) - zbase);

  // 1) Weights -> transposed, padded bf16 (cached in L2 for all 5 steps).
  dim3 tb(32, 8);
  conv_wt<<<dim3(NG_ / 32, KG_ / 32), tb, 0, stream>>>(Wg, Wgt, E_ + H_, 2 * H_);
  conv_wt<<<dim3(NC_ / 32, KG_ / 32), tb, 0, stream>>>(Wc, Wct, E_ + H_, H_);

  // 2) h0 = 0; also zeroes all K/N pad lanes once (never rewritten).
  zero_ws<<<2048, 256, 0, stream>>>((u32x4*)zbase, zbytes / 16);

  // 3) Recurrence.
  for (int t = 0; t < T_; ++t) {
    conv_x<<<dim3((E_ + 255) / 256, B_), 256, 0, stream>>>(inputs, Xb, t);
    gru_gemm<0><<<dim3(NG_ / 128, B_ / 256), 256, 0, stream>>>(
        Xb, Hb, Wgt, bg, Hf, U, RHb, Hb);
    gru_gemm<1><<<dim3(NC_ / 128, B_ / 256), 256, 0, stream>>>(
        Xb, RHb, Wct, bc, Hf, U, RHb, Hb);
  }

  // 4) logits = h @ W_fc + b_fc
  final_fc<<<dim3(B_ / 8), 256, 0, stream>>>(Hf, Wfc, bfc, out);
}
